// MultiHeadAttention_26826365731188
// MI455X (gfx1250) — matrix operand order repaired and checked by reading, not compiled, not run on hardware
//
#include <hip/hip_runtime.h>
#include <hip/hip_bf16.h>

// ---------------------------------------------------------------------------
// MultiHeadAttention forward for MI455X (gfx1250, wave32, WMMA f16->f32)
//   B=2, S=2048, D=1024, H=16, Hd=64
// Pipeline: gemm_bias (QKV) -> flash attention -> gemm_bias (out proj)
// Double-buffered LDS staging: one barrier per K-chunk, global loads issued
// before the math so their latency hides behind WMMA.
// ---------------------------------------------------------------------------

typedef __attribute__((ext_vector_type(16))) _Float16 v16h;
typedef __attribute__((ext_vector_type(8)))  _Float16 v8h;
typedef __attribute__((ext_vector_type(4)))  _Float16 v4h;
typedef __attribute__((ext_vector_type(8)))  float    v8f;

static __device__ __forceinline__ v16h cat8(v8h lo, v8h hi) {
    return __builtin_shufflevector(lo, hi, 0,1,2,3,4,5,6,7,8,9,10,11,12,13,14,15);
}

static __device__ __forceinline__ v4h cvt4(float4 f) {
    v4h r;
    r[0] = (_Float16)f.x; r[1] = (_Float16)f.y;
    r[2] = (_Float16)f.z; r[3] = (_Float16)f.w;
    return r;
}

static __device__ __forceinline__ v8f wmma_f16f32(v16h a, v16h b, v8f c) {
    // D = A(16x32 f16) * B(32x16 f16) + C(16x16 f32)
    return __builtin_amdgcn_wmma_f32_16x16x32_f16(
        /*neg_a=*/false, a, /*neg_b=*/false, b,
        /*c_mod=*/(short)0, c, /*reuse_a=*/false, /*reuse_b=*/false);
}

// ---------------------------------------------------------------------------
// C[M,N] = A[M,K] @ W[K,N] + bias[N]   (A,W,C fp32 row-major; compute in f16)
// Block: 256 threads = 8 waves; WG tile 128x64; wave tile 32x32 (4 WMMAs);
// K chunk 32, double-buffered. Requires M%128==0, N%64==0, K%32==0.
// ---------------------------------------------------------------------------
__global__ __launch_bounds__(256) void mha_gemm_bias(
    const float* __restrict__ A, const float* __restrict__ W,
    const float* __restrict__ bias, float* __restrict__ C,
    int M, int N, int K)
{
    __shared__ __align__(16) _Float16 sA[2][128][32];
    __shared__ __align__(16) _Float16 sB[2][64][32];

    const int tid  = threadIdx.x;
    const int wave = tid >> 5;
    const int lane = tid & 31;
    const int wm   = (wave & 3) << 5;   // 0,32,64,96
    const int wn   = (wave >> 2) << 5;  // 0,32
    const int m0   = blockIdx.x * 128;
    const int n0   = blockIdx.y * 64;

    // fragment coordinates (ISA VGPR layouts)
    const int fm   = lane & 15;         // A-frag row
    const int kb   = (lane >> 4) << 3;  // A-frag k base (0 or 8)
    const int fn   = lane & 15;         // B-frag col
    const int koff = (lane >> 4) << 3;  // B-frag k base (0 or 8)

    // staging coordinates
    const int a_k4 = (tid & 7) * 4;     // 4 consecutive k
    const int a_m  = tid >> 3;          // rows a_m + 32*i
    const int b_n4 = (tid & 15) * 4;    // 4 consecutive n
    const int b_k  = tid >> 4;          // k rows b_k + 16*i

    auto stage_load = [&](int k0, float4 fa[4], float4 fb[2]) {
        #pragma unroll
        for (int i = 0; i < 4; ++i)
            fa[i] = *(const float4*)&A[(size_t)(m0 + a_m + 32 * i) * K + k0 + a_k4];
        #pragma unroll
        for (int i = 0; i < 2; ++i)
            fb[i] = *(const float4*)&W[(size_t)(k0 + b_k + 16 * i) * N + n0 + b_n4];
    };
    auto stage_store = [&](int buf, const float4 fa[4], const float4 fb[2]) {
        #pragma unroll
        for (int i = 0; i < 4; ++i)
            *(v4h*)&sA[buf][a_m + 32 * i][a_k4] = cvt4(fa[i]);
        #pragma unroll
        for (int i = 0; i < 2; ++i) {
            sB[buf][b_n4 + 0][b_k + 16 * i] = (_Float16)fb[i].x;
            sB[buf][b_n4 + 1][b_k + 16 * i] = (_Float16)fb[i].y;
            sB[buf][b_n4 + 2][b_k + 16 * i] = (_Float16)fb[i].z;
            sB[buf][b_n4 + 3][b_k + 16 * i] = (_Float16)fb[i].w;
        }
    };

    v8f acc[2][2];
    #pragma unroll
    for (int mi = 0; mi < 2; ++mi)
        #pragma unroll
        for (int ni = 0; ni < 2; ++ni) acc[mi][ni] = (v8f){};

    float4 fa[4], fb[2];
    stage_load(0, fa, fb);
    stage_store(0, fa, fb);

    const int NC = K >> 5;
    for (int c = 0; c < NC; ++c) {
        __syncthreads();                       // buf[c&1] ready for all waves
        const int cur = c & 1, nxt = cur ^ 1;
        const bool has_next = (c + 1 < NC);

        // Issue next chunk's global loads first; latency hides behind WMMA.
        if (has_next) {
            stage_load((c + 1) * 32, fa, fb);
            if (c + 2 < NC) {                  // L2 prefetch (global_prefetch_b8)
                __builtin_prefetch(&A[(size_t)(m0 + a_m) * K + (c + 2) * 32 + a_k4], 0, 1);
                __builtin_prefetch(&W[(size_t)((c + 2) * 32 + b_k) * N + n0 + b_n4], 0, 1);
            }
        }

        v16h af[2], bf[2];
        #pragma unroll
        for (int mi = 0; mi < 2; ++mi) {
            v8h lo = *(const v8h*)&sA[cur][wm + mi * 16 + fm][kb];
            v8h hi = *(const v8h*)&sA[cur][wm + mi * 16 + fm][kb + 16];
            af[mi] = cat8(lo, hi);
        }
        #pragma unroll
        for (int ni = 0; ni < 2; ++ni) {
            v8h lo = *(const v8h*)&sB[cur][wn + ni * 16 + fn][koff];
            v8h hi = *(const v8h*)&sB[cur][wn + ni * 16 + fn][koff + 16];
            bf[ni] = cat8(lo, hi);
        }
        #pragma unroll
        for (int mi = 0; mi < 2; ++mi)
            #pragma unroll
            for (int ni = 0; ni < 2; ++ni)
                acc[mi][ni] = wmma_f16f32(af[mi], bf[ni], acc[mi][ni]);

        if (has_next) stage_store(nxt, fa, fb);
    }

    // C/D layout: lane n = lane&15; VGPR r <-> M = r + 8*(lane>>4)
    const int rowbase = (lane >> 4) << 3;
    #pragma unroll
    for (int mi = 0; mi < 2; ++mi) {
        #pragma unroll
        for (int ni = 0; ni < 2; ++ni) {
            float bia = bias[n0 + wn + ni * 16 + fn];
            #pragma unroll
            for (int r = 0; r < 8; ++r) {
                int m = m0 + wm + mi * 16 + rowbase + r;
                C[(size_t)m * N + n0 + wn + ni * 16 + fn] = acc[mi][ni][r] + bia;
            }
        }
    }
}

// ---------------------------------------------------------------------------
// Flash attention: grid (S/64, B*H), 128 threads = 4 waves.
// Each wave owns a 16-row query tile; the block shares double-buffered K/V.
// qkv: [B, S, 3*D] fp32 ; ctx: [B, S, D] fp32
// ---------------------------------------------------------------------------
__global__ __launch_bounds__(128) void mha_flash_attn(
    const float* __restrict__ qkv, float* __restrict__ ctx, int B, int S)
{
    constexpr int D = 1024, H = 16, HD = 64;
    constexpr int RS = 3 * D;                 // floats per token in qkv
    __shared__ __align__(16) _Float16 sQ[64][64];         // q rows (per-block)
    __shared__ __align__(16) _Float16 sK[2][32][64];      // [key][dim]
    __shared__ __align__(16) _Float16 sVT[2][64][32];     // [dim][key]
    __shared__ __align__(16) _Float16 sP[4][16][32];      // per-wave P transpose

    const int tid  = threadIdx.x;
    const int wave = tid >> 5;
    const int lane = tid & 31;
    const int b    = blockIdx.y / H;
    const int h    = blockIdx.y % H;
    const int q0   = blockIdx.x * 64;

    const float* base = qkv + (size_t)b * S * RS;

    // Stage Q (64 rows x 64 dims) once; float4 -> v4h
    #pragma unroll
    for (int i = 0; i < 8; ++i) {
        int q  = i * 128 + tid;
        int d4 = q & 15, r = q >> 4;
        float4 f = *(const float4*)&base[(size_t)(q0 + r) * RS + (size_t)h * HD + d4 * 4];
        *(v4h*)&sQ[r][d4 * 4] = cvt4(f);
    }
    __syncthreads();

    const int fm   = lane & 15;
    const int kb   = (lane >> 4) << 3;
    const int fn   = lane & 15;
    const int koff = (lane >> 4) << 3;

    // Q fragments: dims [0,32) and [32,64) are the WMMA K dimension
    v16h qf[2];
    #pragma unroll
    for (int c = 0; c < 2; ++c) {
        v8h lo = *(const v8h*)&sQ[wave * 16 + fm][c * 32 + kb];
        v8h hi = *(const v8h*)&sQ[wave * 16 + fm][c * 32 + kb + 16];
        qf[c] = cat8(lo, hi);
    }

    // KV staging coordinates
    const int s_d4 = (tid & 15) * 4;
    const int s_r  = tid >> 4;               // rows s_r + 8*i

    auto kv_load = [&](int kt, float4 fk[4], float4 fv[4]) {
        #pragma unroll
        for (int i = 0; i < 4; ++i) {
            const float* rowp = &base[(size_t)(kt + s_r + 8 * i) * RS + (size_t)h * HD + s_d4];
            fk[i] = *(const float4*)(rowp + D);
            fv[i] = *(const float4*)(rowp + 2 * D);
        }
    };
    auto kv_store = [&](int buf, const float4 fk[4], const float4 fv[4]) {
        #pragma unroll
        for (int i = 0; i < 4; ++i) {
            *(v4h*)&sK[buf][s_r + 8 * i][s_d4] = cvt4(fk[i]);
            sVT[buf][s_d4 + 0][s_r + 8 * i] = (_Float16)fv[i].x;
            sVT[buf][s_d4 + 1][s_r + 8 * i] = (_Float16)fv[i].y;
            sVT[buf][s_d4 + 2][s_r + 8 * i] = (_Float16)fv[i].z;
            sVT[buf][s_d4 + 3][s_r + 8 * i] = (_Float16)fv[i].w;
        }
    };

    float mrun[8], lrun[8];
    v8f acc[4];
    #pragma unroll
    for (int r = 0; r < 8; ++r) { mrun[r] = -1e30f; lrun[r] = 0.0f; }
    #pragma unroll
    for (int j = 0; j < 4; ++j) acc[j] = (v8f){};

    const float scale = 0.125f;  // 1/sqrt(64)

    float4 fk[4], fv[4];
    kv_load(0, fk, fv);
    kv_store(0, fk, fv);

    const int NT = S >> 5;
    for (int t = 0; t < NT; ++t) {
        __syncthreads();                     // buf[t&1] ready for all waves
        const int cur = t & 1, nxt = cur ^ 1;
        const bool has_next = (t + 1 < NT);

        if (has_next) {
            kv_load((t + 1) * 32, fk, fv);
            if (t + 2 < NT) {                // L2 prefetch (global_prefetch_b8)
                const float* rowp = &base[(size_t)((t + 2) * 32 + s_r) * RS + (size_t)h * HD + s_d4];
                __builtin_prefetch(rowp + D, 0, 1);
                __builtin_prefetch(rowp + 2 * D, 0, 1);
            }
        }

        // Scores: s0 = q @ K[0:16]^T, s1 = q @ K[16:32]^T  (K-dim = 64, 2 chunks)
        v8f s0 = {}; v8f s1 = {};
        #pragma unroll
        for (int c = 0; c < 2; ++c) {
            v8h l0 = *(const v8h*)&sK[cur][fn][c * 32 + koff];
            v8h h0 = *(const v8h*)&sK[cur][fn][c * 32 + koff + 16];
            s0 = wmma_f16f32(qf[c], cat8(l0, h0), s0);
            v8h l1 = *(const v8h*)&sK[cur][16 + fn][c * 32 + koff];
            v8h h1 = *(const v8h*)&sK[cur][16 + fn][c * 32 + koff + 16];
            s1 = wmma_f16f32(qf[c], cat8(l1, h1), s1);
        }

        // Online softmax over the 32 new keys per row.
        // Row r lives across a 16-lane half; xor masks 1..8 stay within a half.
        float p0[8], p1[8];
        #pragma unroll
        for (int r = 0; r < 8; ++r) {
            float a0 = s0[r] * scale, a1 = s1[r] * scale;
            float mx = fmaxf(a0, a1);
            #pragma unroll
            for (int off = 1; off < 16; off <<= 1) mx = fmaxf(mx, __shfl_xor(mx, off));
            float mnew = fmaxf(mrun[r], mx);
            float e0 = __expf(a0 - mnew);
            float e1 = __expf(a1 - mnew);
            float rs = e0 + e1;
            #pragma unroll
            for (int off = 1; off < 16; off <<= 1) rs += __shfl_xor(rs, off);
            float alpha = __expf(mrun[r] - mnew);
            lrun[r] = lrun[r] * alpha + rs;
            mrun[r] = mnew;
            p0[r] = e0; p1[r] = e1;
            #pragma unroll
            for (int j = 0; j < 4; ++j) acc[j][r] *= alpha;
        }

        // Transpose P (C layout) -> A-fragment layout via wave-private LDS
        // buffer. Same-wave LDS ops are in-order (DScnt), so no barrier needed.
        const int rowbase = (lane >> 4) << 3;
        #pragma unroll
        for (int r = 0; r < 8; ++r) {
            sP[wave][rowbase + r][fn]      = (_Float16)p0[r];
            sP[wave][rowbase + r][16 + fn] = (_Float16)p1[r];
        }
        v8h plo = *(const v8h*)&sP[wave][fm][kb];
        v8h phi = *(const v8h*)&sP[wave][fm][kb + 16];
        v16h pf = cat8(plo, phi);

        // acc[j] += P[16x32] @ V[32 x 16dims(j)]
        #pragma unroll
        for (int j = 0; j < 4; ++j) {
            v8h vl = *(const v8h*)&sVT[cur][j * 16 + fn][koff];
            v8h vh = *(const v8h*)&sVT[cur][j * 16 + fn][koff + 16];
            acc[j] = wmma_f16f32(pf, cat8(vl, vh), acc[j]);
        }

        if (has_next) kv_store(nxt, fk, fv);
    }

    // Epilogue: divide by l, scatter into ctx[b, s, h*64 + d]
    float* out = ctx + (size_t)b * S * D;
    const int rowbase = (lane >> 4) << 3;
    #pragma unroll
    for (int r = 0; r < 8; ++r) {
        float inv = 1.0f / lrun[r];
        int m = q0 + wave * 16 + rowbase + r;
        #pragma unroll
        for (int j = 0; j < 4; ++j) {
            out[(size_t)m * D + (size_t)h * HD + j * 16 + fn] = acc[j][r] * inv;
        }
    }
}

// ---------------------------------------------------------------------------
extern "C" void kernel_launch(void* const* d_in, const int* in_sizes, int n_in,
                              void* d_out, int out_size, void* d_ws, size_t ws_size,
                              hipStream_t stream) {
    (void)in_sizes; (void)n_in; (void)out_size; (void)ws_size;
    const float* x     = (const float*)d_in[0];   // [B,S,D]
    const float* W_qkv = (const float*)d_in[1];   // [D,3D]
    const float* b_qkv = (const float*)d_in[2];   // [3D]
    const float* W_out = (const float*)d_in[3];   // [D,D]
    const float* b_out = (const float*)d_in[4];   // [D]
    float* out = (float*)d_out;                   // [B,S,D]

    const int B = 2, S = 2048, D = 1024, H = 16;
    const int M = B * S;                          // 4096

    float* qkv = (float*)d_ws;                    // [B,S,3D]  50.3 MB
    float* ctx = qkv + (size_t)M * 3 * D;         // [B,S,D]   16.8 MB

    // 1) QKV projection: [4096,1024] @ [1024,3072] + bias
    dim3 blk(256);
    dim3 g1(M / 128, (3 * D) / 64);
    mha_gemm_bias<<<g1, blk, 0, stream>>>(x, W_qkv, b_qkv, qkv, M, 3 * D, D);

    // 2) Flash attention per (b,h), 64 query rows per block
    dim3 g2(S / 64, B * H);
    mha_flash_attn<<<g2, dim3(128), 0, stream>>>(qkv, ctx, B, S);

    // 3) Output projection: [4096,1024] @ [1024,1024] + bias
    dim3 g3(M / 128, D / 64);
    mha_gemm_bias<<<g3, blk, 0, stream>>>(ctx, W_out, b_out, out, M, D, D);
}
